// RecurrentTemporalDecoder_42494406426983
// MI455X (gfx1250) — matrix operand order repaired and checked by reading, not compiled, not run on hardware
//
#include <hip/hip_runtime.h>
#include <hip/hip_bf16.h>

#define NCAM   15
#define H      1024
#define G3     3072          // 3*H
#define BATCH  256
#define TSTEPS 60
#define MAXTILES 32          // max padded 16-row M-tiles: sum ceil(cnt_c/16) <= 30
#define KSLICES 2            // split-K factor (deterministic two-term sum in update)
#define KSLICE_LEN (H / KSLICES)   // 512

typedef __bf16 bf16_t;
typedef __attribute__((ext_vector_type(8)))  __bf16 v8bf;
typedef __attribute__((ext_vector_type(16))) __bf16 v16bf;
typedef __attribute__((ext_vector_type(8)))  float  v8f;

#if defined(__has_builtin)
#if __has_builtin(__builtin_amdgcn_sched_barrier)
#define SCHED_FENCE() __builtin_amdgcn_sched_barrier(0)
#endif
#endif
#ifndef SCHED_FENCE
#define SCHED_FENCE() ((void)0)
#endif

static __device__ inline unsigned short f2bf(float f) {
    unsigned int u = __float_as_uint(f);
    unsigned int r = (u + 0x7FFFu + ((u >> 16) & 1u)) >> 16;   // round-to-nearest-even
    return (unsigned short)r;
}

static __device__ inline v16bf load_a_frag(const bf16_t* p0, const bf16_t* p1) {
    v8bf a = *(const v8bf*)p0;   // 16B aligned
    v8bf b = *(const v8bf*)p1;
    v16bf r;
#pragma unroll
    for (int i = 0; i < 8; ++i) { r[i] = a[i]; r[i + 8] = b[i]; }
    return r;
}

// ---------------------------------------------------------------------------
// Group samples by camera into 16-row M-tiles (padded rows = -1).
// ---------------------------------------------------------------------------
__global__ __launch_bounds__(256) void bucket_kernel(const int* __restrict__ cam,
                                                     int* __restrict__ tile_cam,
                                                     int* __restrict__ row2s) {
    __shared__ int cnt[NCAM];
    __shared__ int off[NCAM];
    __shared__ int tstart[NCAM];
    int t = threadIdx.x;
    if (t < NCAM) cnt[t] = 0;
    __syncthreads();
    int c = cam[t];
    atomicAdd(&cnt[c], 1);
    // init row2s to -1 (512 entries)
    row2s[t] = -1;
    row2s[t + 256] = -1;
    __syncthreads();
    if (t == 0) {
        int tiles = 0;
        for (int i = 0; i < NCAM; ++i) {
            tstart[i] = tiles;
            int nt = (cnt[i] + 15) >> 4;
            for (int j = 0; j < nt; ++j) tile_cam[tiles + j] = i;
            tiles += nt;
        }
        for (int j = tiles; j < MAXTILES; ++j) tile_cam[j] = -1;
    }
    if (t < NCAM) off[t] = 0;
    __syncthreads();
    int pos = atomicAdd(&off[c], 1);
    row2s[tstart[c] * 16 + pos] = t;
}

// ---------------------------------------------------------------------------
// fp32 -> bf16 weight conversion (one-time; makes W_hh 90MB => L2-resident)
// ---------------------------------------------------------------------------
__global__ __launch_bounds__(256) void cvt_kernel(const float* __restrict__ src,
                                                  unsigned short* __restrict__ dst,
                                                  int n4) {
    int i = (blockIdx.x * 256 + threadIdx.x);
    if (i < n4) {
        float4 v = ((const float4*)src)[i];
        unsigned short* d = dst + (size_t)i * 4;
        d[0] = f2bf(v.x); d[1] = f2bf(v.y); d[2] = f2bf(v.z); d[3] = f2bf(v.w);
    }
}

// ---------------------------------------------------------------------------
// One-time input projection: gi[b,g] = x[b,:] . W_ih[cam[b],g,:] + b_ih[cam[b],g]
// ---------------------------------------------------------------------------
__global__ __launch_bounds__(256) void gi_kernel(const float* __restrict__ x,
                                                 const int* __restrict__ cam,
                                                 const float* __restrict__ Wih,
                                                 const float* __restrict__ bih,
                                                 float* __restrict__ Gi) {
    int wave = threadIdx.x >> 5;
    int lane = threadIdx.x & 31;
    int g = blockIdx.x * 8 + wave;        // gridDim.x = G3/8
    int b = blockIdx.y;                   // gridDim.y = BATCH
    int c = cam[b];
    const float* w  = Wih + ((size_t)c * G3 + g) * H;
    const float* xr = x + (size_t)b * H;
    float s = 0.f;
    for (int k = lane * 4; k < H; k += 32 * 4) {
        float4 wv = *(const float4*)(w + k);
        float4 xv = *(const float4*)(xr + k);
        s += wv.x * xv.x + wv.y * xv.y + wv.z * xv.z + wv.w * xv.w;
    }
#pragma unroll
    for (int o = 16; o > 0; o >>= 1) s += __shfl_down(s, o, 32);
    if (lane == 0) Gi[(size_t)b * G3 + g] = s + bih[(size_t)c * G3 + g];
}

__global__ __launch_bounds__(256) void init_kernel(float* __restrict__ h,
                                                   unsigned short* __restrict__ hbf) {
    int idx = blockIdx.x * 256 + threadIdx.x;
    h[idx] = 0.f;
    hbf[idx] = 0;
}

// ---------------------------------------------------------------------------
// Per-step hidden GEMM: gh_partial[kslice] = h_bf16[:, kslice] @ W_hh[cam]^T slice
// Block = 4 waves; each wave computes one 16x64 output tile with
// v_wmma_f32_16x16x32_bf16 over a 512-wide K slice (16 k-iterations).
// Software pipelined: sched_barriers force iteration i+1's load clause to
// issue BEFORE iteration i's WMMAs, so the waitcnt pass waits only on the
// older load group (steady-state s_wait_loadcnt ~10, not 0). Occupancy here
// is only ~2 waves/SIMD, so in-wave latency hiding is what matters.
// ---------------------------------------------------------------------------
__global__ __launch_bounds__(128) void gemm_gh(const bf16_t* __restrict__ Wbf,  // [NCAM,G3,H]
                                               const bf16_t* __restrict__ Abf,  // [BATCH,H]
                                               float* __restrict__ GhP,         // [KSLICES,BATCH,G3]
                                               const int* __restrict__ tile_cam,
                                               const int* __restrict__ row2s) {
    int tile = blockIdx.x;
    int cdat = tile_cam[tile];
    if (cdat < 0) return;                         // wave-uniform: EXEC stays all-ones
    int wave   = threadIdx.x >> 5;
    int lane   = threadIdx.x & 31;
    int lane16 = lane & 15;
    int hi     = lane >> 4;
    int n0     = blockIdx.y * 256 + wave * 64;    // this wave's 64 output columns
    int kbase  = blockIdx.z * KSLICE_LEN;         // this block's K slice

    int srow = row2s[tile * 16 + lane16];
    int arow = (srow < 0) ? 0 : srow;             // padded rows compute garbage, masked at store
    const bf16_t* aptr  = Abf + (size_t)arow * H + hi * 16 + kbase;
    const bf16_t* bbase = Wbf + (size_t)cdat * G3 * H + (size_t)(n0 + lane16) * H + hi * 16 + kbase;

    v8f acc0 = {}; v8f acc1 = {}; v8f acc2 = {}; v8f acc3 = {};

    // ---- prologue: fetch k-iteration 0 ----
    v16bf a  = load_a_frag(aptr, aptr + 8);
    v16bf b0 = *(const v16bf*)(bbase);
    v16bf b1 = *(const v16bf*)(bbase + (size_t)16 * H);
    v16bf b2 = *(const v16bf*)(bbase + (size_t)32 * H);
    v16bf b3 = *(const v16bf*)(bbase + (size_t)48 * H);

    // ---- pipelined main loop: issue loads for k0+32, fence, then WMMA k0 ----
#pragma unroll
    for (int k0 = 0; k0 < KSLICE_LEN - 32; k0 += 32) {
        int kn = k0 + 32;
        v16bf an  = load_a_frag(aptr + kn, aptr + kn + 8);
        v16bf bn0 = *(const v16bf*)(bbase + kn);
        v16bf bn1 = *(const v16bf*)(bbase + (size_t)16 * H + kn);
        v16bf bn2 = *(const v16bf*)(bbase + (size_t)32 * H + kn);
        v16bf bn3 = *(const v16bf*)(bbase + (size_t)48 * H + kn);
        SCHED_FENCE();   // next-group loads must be in flight before these WMMAs
        acc0 = __builtin_amdgcn_wmma_f32_16x16x32_bf16(false, a, false, b0, (short)0, acc0, false, false);
        acc1 = __builtin_amdgcn_wmma_f32_16x16x32_bf16(false, a, false, b1, (short)0, acc1, false, false);
        acc2 = __builtin_amdgcn_wmma_f32_16x16x32_bf16(false, a, false, b2, (short)0, acc2, false, false);
        acc3 = __builtin_amdgcn_wmma_f32_16x16x32_bf16(false, a, false, b3, (short)0, acc3, false, false);
        SCHED_FENCE();   // keep the WMMAs from drifting into the next load clause
        a = an; b0 = bn0; b1 = bn1; b2 = bn2; b3 = bn3;
    }
    // ---- epilogue ----
    acc0 = __builtin_amdgcn_wmma_f32_16x16x32_bf16(false, a, false, b0, (short)0, acc0, false, false);
    acc1 = __builtin_amdgcn_wmma_f32_16x16x32_bf16(false, a, false, b1, (short)0, acc1, false, false);
    acc2 = __builtin_amdgcn_wmma_f32_16x16x32_bf16(false, a, false, b2, (short)0, acc2, false, false);
    acc3 = __builtin_amdgcn_wmma_f32_16x16x32_bf16(false, a, false, b3, (short)0, acc3, false, false);

    // D layout: vgpr r -> (M = r + 8*hi, N = lane16)
    float* GhS = GhP + (size_t)blockIdx.z * BATCH * G3;
#pragma unroll
    for (int r = 0; r < 8; ++r) {
        int rowm = r + hi * 8;
        int s = row2s[tile * 16 + rowm];
        if (s >= 0) {
            float* o = GhS + (size_t)s * G3 + n0 + lane16;
            o[0]  = acc0[r];
            o[16] = acc1[r];
            o[32] = acc2[r];
            o[48] = acc3[r];
        }
    }
}

// ---------------------------------------------------------------------------
// GRU elementwise update; sums the KSLICES partial GEMM results in a fixed
// order (bit-deterministic), applies gates, writes h (fp32), h (bf16 for the
// next GEMM), and out[b,h,t].
// ---------------------------------------------------------------------------
__global__ __launch_bounds__(256) void update_kernel(const float* __restrict__ Gi,
                                                     const float* __restrict__ GhP,
                                                     const float* __restrict__ bhh,
                                                     const int* __restrict__ cam,
                                                     float* __restrict__ h,
                                                     unsigned short* __restrict__ hbf,
                                                     float* __restrict__ out,
                                                     int t) {
    int idx = blockIdx.x * 256 + threadIdx.x;     // BATCH*H threads
    int b = idx >> 10;
    int j = idx & (H - 1);
    int c = cam[b];
    const float* gi  = Gi + (size_t)b * G3;
    const float* gh0 = GhP + (size_t)b * G3;
    const float* gh1 = GhP + (size_t)BATCH * G3 + (size_t)b * G3;
    const float* bh  = bhh + (size_t)c * G3;
    float gir = gi[j], giz = gi[H + j], gin = gi[2 * H + j];
    float ghr = (gh0[j]         + gh1[j])         + bh[j];
    float ghz = (gh0[H + j]     + gh1[H + j])     + bh[H + j];
    float ghn = (gh0[2 * H + j] + gh1[2 * H + j]) + bh[2 * H + j];
    float r = 1.f / (1.f + __expf(-(gir + ghr)));
    float z = 1.f / (1.f + __expf(-(giz + ghz)));
    float n = tanhf(gin + r * ghn);
    float hp = h[idx];
    float hn = (1.f - z) * n + z * hp;
    h[idx] = hn;
    hbf[idx] = f2bf(hn);
    out[((size_t)b * H + j) * TSTEPS + t] = hn;
}

// ---------------------------------------------------------------------------
extern "C" void kernel_launch(void* const* d_in, const int* in_sizes, int n_in,
                              void* d_out, int out_size, void* d_ws, size_t ws_size,
                              hipStream_t stream) {
    const float* x    = (const float*)d_in[0];
    const int*   cam  = (const int*)d_in[1];
    const float* Wih  = (const float*)d_in[2];
    const float* Whh  = (const float*)d_in[3];
    const float* bih  = (const float*)d_in[4];
    const float* bhh  = (const float*)d_in[5];
    float* out = (float*)d_out;

    // workspace layout (all chunk sizes are multiples of 128B)
    char* p = (char*)d_ws;
    unsigned short* whh_bf = (unsigned short*)p; p += (size_t)NCAM * G3 * H * 2;     // 90 MB
    float* gi  = (float*)p;            p += (size_t)BATCH * G3 * 4;                  // 3 MB
    float* ghp = (float*)p;            p += (size_t)KSLICES * BATCH * G3 * 4;        // 6 MB
    float* h   = (float*)p;            p += (size_t)BATCH * H * 4;                   // 1 MB
    unsigned short* hbf = (unsigned short*)p; p += (size_t)BATCH * H * 2;            // 0.5 MB
    int* tile_cam = (int*)p;           p += MAXTILES * 4;
    int* row2s    = (int*)p;           p += MAXTILES * 16 * 4;

    // 1) group samples by camera into WMMA M-tiles
    bucket_kernel<<<1, 256, 0, stream>>>(cam, tile_cam, row2s);

    // 2) convert W_hh to bf16 once (L2-resident thereafter)
    {
        int n4 = NCAM * G3 * H / 4;
        cvt_kernel<<<(n4 + 255) / 256, 256, 0, stream>>>(Whh, whh_bf, n4);
    }

    // 3) one-time input projection gi = x @ W_ih[cam]^T + b_ih[cam]
    gi_kernel<<<dim3(G3 / 8, BATCH), 256, 0, stream>>>(x, cam, Wih, bih, gi);

    // 4) h0 = 0
    init_kernel<<<(BATCH * H) / 256, 256, 0, stream>>>(h, hbf);

    // 5) recurrence: 60 x (split-K WMMA GEMM + GRU update)
    const bf16_t* whh_t = (const bf16_t*)whh_bf;
    const bf16_t* hbf_t = (const bf16_t*)hbf;
    for (int t = 0; t < TSTEPS; ++t) {
        gemm_gh<<<dim3(MAXTILES, G3 / 256, KSLICES), 128, 0, stream>>>(whh_t, hbf_t, ghp, tile_cam, row2s);
        update_kernel<<<(BATCH * H) / 256, 256, 0, stream>>>(gi, ghp, bhh, cam, h, hbf, out, t);
    }
}
